// SelfAttention1D_63256278335496
// MI455X (gfx1250) — matrix operand order repaired and checked by reading, not compiled, not run on hardware
//
#include <hip/hip_runtime.h>
#include <hip/hip_bf16.h>

typedef _Float16 half_t;
typedef __attribute__((ext_vector_type(16))) _Float16 v16h;
typedef __attribute__((ext_vector_type(8)))  _Float16 v8h;
typedef __attribute__((ext_vector_type(8)))  float    v8f;
typedef __attribute__((ext_vector_type(4)))  int      v4i;

#define BB 4
#define SS 2048
#define CC 512
#define HH 8
#define GG 32
#define GC 16            // C / G
#define DD 64            // C / H
#define EPS 1e-3f
#define MT (BB*SS)       // 8192 rows

#define TM 128
#define TN 128
#define TK 32

// -------- CDNA5 async global->LDS copy (ASYNCcnt-tracked), with fallback ----
#if defined(__gfx1250__) && \
    __has_builtin(__builtin_amdgcn_global_load_async_to_lds_b128) && \
    __has_builtin(__builtin_amdgcn_s_wait_asynccnt)
#define USE_ASYNC_LDS 1
#else
#define USE_ASYNC_LDS 0
#endif

__device__ __forceinline__ void cp_glds16(void* lds_dst, const void* gsrc) {
#if USE_ASYNC_LDS
    __builtin_amdgcn_global_load_async_to_lds_b128(
        (__attribute__((address_space(1))) v4i*)(gsrc),
        (__attribute__((address_space(3))) v4i*)(lds_dst), 0, 0);
#else
    *(v8h*)lds_dst = *(const v8h*)gsrc;
#endif
}

__device__ __forceinline__ void cp_glds_wait() {
#if USE_ASYNC_LDS
    __builtin_amdgcn_s_wait_asynccnt(0);
#endif
}

__device__ __forceinline__ v8f wmma_f16(v16h a, v16h b, v8f c) {
    return __builtin_amdgcn_wmma_f32_16x16x32_f16(
        false, a, false, b, (short)0, c, false, false);
}

// ---------------------------------------------------------------- stats
__global__ void gn_stats_kernel(const float* __restrict__ x,
                                float* __restrict__ stats /*[B*G][2]*/) {
    int bg = blockIdx.x;              // B*G = 128
    int b = bg / GG, g = bg % GG;
    const float* xp = x + (size_t)b * SS * CC + g * GC;
    int tid = threadIdx.x;
    float sum = 0.f, sq = 0.f;
    for (int s = tid; s < SS; s += blockDim.x) {
        const float4* row = (const float4*)(xp + (size_t)s * CC);
        #pragma unroll
        for (int j = 0; j < GC / 4; ++j) {
            float4 v = row[j];
            sum += v.x + v.y + v.z + v.w;
            sq  += v.x*v.x + v.y*v.y + v.z*v.z + v.w*v.w;
        }
    }
    __shared__ float rs[256], rq[256];
    rs[tid] = sum; rq[tid] = sq;
    __syncthreads();
    for (int off = 128; off > 0; off >>= 1) {
        if (tid < off) { rs[tid] += rs[tid + off]; rq[tid] += rq[tid + off]; }
        __syncthreads();
    }
    if (tid == 0) {
        float inv = 1.f / (float)(SS * GC);
        float mean = rs[0] * inv;
        float var  = rq[0] * inv - mean * mean;
        stats[bg * 2 + 0] = mean;
        stats[bg * 2 + 1] = rsqrtf(var + EPS);
    }
}

// ---------------------------------------------------------------- normalize -> f16
__global__ void gn_apply_kernel(const float* __restrict__ x,
                                const float* __restrict__ gamma,
                                const float* __restrict__ beta,
                                const float* __restrict__ stats,
                                half_t* __restrict__ xn) {
    size_t i = ((size_t)blockIdx.x * blockDim.x + threadIdx.x) * 4;  // B*S*C total
    int c = (int)(i % CC);
    size_t bs = i / CC;
    int b = (int)(bs / SS);
    int g = c / GC;
    const float* st = stats + (size_t)(b * GG + g) * 2;
    float mean = st[0], rstd = st[1];
    float4 xv = *(const float4*)(x + i);
    xn[i + 0] = (half_t)((xv.x - mean) * rstd * gamma[c + 0] + beta[c + 0]);
    xn[i + 1] = (half_t)((xv.y - mean) * rstd * gamma[c + 1] + beta[c + 1]);
    xn[i + 2] = (half_t)((xv.z - mean) * rstd * gamma[c + 2] + beta[c + 2]);
    xn[i + 3] = (half_t)((xv.w - mean) * rstd * gamma[c + 3] + beta[c + 3]);
}

// ---------------------------------------------------------------- weight transpose fp32[K][N] -> f16[N][K]
__global__ void wt_convert_kernel(const float* __restrict__ w, half_t* __restrict__ wT,
                                  int K, int N) {
    int idx = blockIdx.x * blockDim.x + threadIdx.x;
    if (idx < K * N) {
        int k = idx / N, n = idx % N;
        wT[(size_t)n * K + k] = (half_t)w[idx];
    }
}

// ---------------------------------------------------------------- QKV GEMM: [M,512] x [512,1536]
__global__ __launch_bounds__(256)
void qkv_gemm_kernel(const half_t* __restrict__ A,     // [M][512]
                     const half_t* __restrict__ Bt,    // [1536][512]
                     half_t* __restrict__ qh, half_t* __restrict__ kh,
                     half_t* __restrict__ vh) {
    __shared__ half_t sA[TM * TK];
    __shared__ half_t sB[TN * TK];
    const int tid = threadIdx.x, wave = tid >> 5, lane = tid & 31;
    const int wm = wave & 3, wn = wave >> 2;
    const int m0 = blockIdx.x * TM, n0 = blockIdx.y * TN;
    const int lr = tid >> 1, lk = (tid & 1) * 16;
    const int r = lane & 15, hi = lane >> 4;

    const v8f vz = {0.f,0.f,0.f,0.f,0.f,0.f,0.f,0.f};
    v8f acc[2][4];
    #pragma unroll
    for (int i = 0; i < 2; ++i)
        #pragma unroll
        for (int j = 0; j < 4; ++j) acc[i][j] = vz;

    for (int k0 = 0; k0 < CC; k0 += TK) {
        const half_t* ga = A  + (size_t)(m0 + lr) * CC + k0 + lk;
        const half_t* gb = Bt + (size_t)(n0 + lr) * CC + k0 + lk;
        __builtin_prefetch(ga + TK, 0, 0);
        __builtin_prefetch(gb + TK, 0, 0);
        cp_glds16(sA + lr * TK + lk,     ga);
        cp_glds16(sA + lr * TK + lk + 8, ga + 8);
        cp_glds16(sB + lr * TK + lk,     gb);
        cp_glds16(sB + lr * TK + lk + 8, gb + 8);
        cp_glds_wait();
        __syncthreads();

        v16h af[2], bf[4];
        #pragma unroll
        for (int i = 0; i < 2; ++i) {
            const half_t* p = sA + (wm * 32 + i * 16 + r) * TK + hi * 16;
            union { v16h v; v8h h[2]; } u;
            u.h[0] = *(const v8h*)p;
            u.h[1] = *(const v8h*)(p + 8);
            af[i] = u.v;
        }
        #pragma unroll
        for (int j = 0; j < 4; ++j)
            bf[j] = *(const v16h*)(sB + (wn * 64 + j * 16 + r) * TK + hi * 16);
        #pragma unroll
        for (int i = 0; i < 2; ++i)
            #pragma unroll
            for (int j = 0; j < 4; ++j)
                acc[i][j] = wmma_f16(af[i], bf[j], acc[i][j]);
        __syncthreads();
    }

    // scatter to q/k/v [B,H,S,D] f16; q pre-scaled by 1/sqrt(D)
    #pragma unroll
    for (int i = 0; i < 2; ++i) {
        #pragma unroll
        for (int j = 0; j < 4; ++j) {
            int n = n0 + wn * 64 + j * 16 + r;
            int s3 = n >> 9;             // 0:q 1:k 2:v
            int h  = (n >> 6) & 7;
            int d  = n & 63;
            float scl = (s3 == 0) ? 0.125f : 1.0f;
            half_t* dst = (s3 == 0) ? qh : ((s3 == 1) ? kh : vh);
            #pragma unroll
            for (int v = 0; v < 8; ++v) {
                int m = m0 + wm * 32 + i * 16 + v + hi * 8;
                int b = m >> 11, s = m & 2047;
                dst[(((size_t)(b * HH + h) * SS + s) << 6) + d] =
                    (half_t)(acc[i][j][v] * scl);
            }
        }
    }
}

// ---------------------------------------------------------------- flash attention
#define AKV 64
__global__ __launch_bounds__(256)
void attn_kernel(const half_t* __restrict__ qh, const half_t* __restrict__ kh,
                 const half_t* __restrict__ vh, half_t* __restrict__ oh) {
    __shared__ half_t sK[AKV * DD];          // [key][d]
    __shared__ half_t sVt[DD * AKV];         // [d][key]
    __shared__ half_t sP[8][16 * AKV];       // per-wave scores / probs
    __shared__ float  sM[8][16], sL[8][16], sAl[8][16];

    const int bh = blockIdx.y;               // b*H + h
    const int tid = threadIdx.x, wave = tid >> 5, lane = tid & 31;
    const int r = lane & 15, hi = lane >> 4;
    const int q0 = blockIdx.x * 128 + wave * 16;
    const half_t* qp = qh + (size_t)bh * SS * DD;
    const half_t* kp = kh + (size_t)bh * SS * DD;
    const half_t* vp = vh + (size_t)bh * SS * DD;

    // Q fragments (held entire loop)
    v16h qf[2];
    #pragma unroll
    for (int st = 0; st < 2; ++st) {
        const half_t* p = qp + (size_t)(q0 + r) * DD + st * 32 + hi * 16;
        union { v16h v; v8h h[2]; } u;
        u.h[0] = *(const v8h*)p;
        u.h[1] = *(const v8h*)(p + 8);
        qf[st] = u.v;
    }
    const v8f vz = {0.f,0.f,0.f,0.f,0.f,0.f,0.f,0.f};
    v8f o[4];
    #pragma unroll
    for (int j = 0; j < 4; ++j) o[j] = vz;
    if (lane < 16) { sM[wave][lane] = -1e30f; sL[wave][lane] = 0.f; }

    for (int kb = 0; kb < SS; kb += AKV) {
        { // cooperative staging: K row-major (async direct-to-LDS), V transposed
            int key = tid >> 2;
            int dc  = (tid & 3) * 16;
            const half_t* gk = kp + (size_t)(kb + key) * DD + dc;
            cp_glds16(sK + key * DD + dc,     gk);
            cp_glds16(sK + key * DD + dc + 8, gk + 8);
            union { v8h h8[2]; half_t e[16]; } uv;
            const v8h* gv = (const v8h*)(vp + (size_t)(kb + key) * DD + dc);
            uv.h8[0] = gv[0]; uv.h8[1] = gv[1];
            #pragma unroll
            for (int jj = 0; jj < 16; ++jj) sVt[(dc + jj) * AKV + key] = uv.e[jj];
        }
        cp_glds_wait();
        __syncthreads();

        // scores S = Q K^T (q pre-scaled), spill to LDS as f16
        #pragma unroll
        for (int kt = 0; kt < 4; ++kt) {
            v8f s = vz;
            #pragma unroll
            for (int st = 0; st < 2; ++st) {
                v16h bf = *(const v16h*)(sK + (kt * 16 + r) * DD + st * 32 + hi * 16);
                s = wmma_f16(qf[st], bf, s);
            }
            #pragma unroll
            for (int v = 0; v < 8; ++v)
                sP[wave][(v + hi * 8) * AKV + kt * 16 + r] = (half_t)s[v];
        }
        __syncthreads();

        { // online softmax: lane pair (r, r+16) handles row r
            int c0 = hi * 32;
            float vals[32];
            float mx = -1e30f;
            #pragma unroll
            for (int c = 0; c < 32; ++c) {
                vals[c] = (float)sP[wave][r * AKV + c0 + c];
                mx = fmaxf(mx, vals[c]);
            }
            mx = fmaxf(mx, __shfl_xor(mx, 16, 32));
            float mOld = sM[wave][r];
            float mNew = fmaxf(mOld, mx);
            float ls = 0.f;
            #pragma unroll
            for (int c = 0; c < 32; ++c) {
                float p = __expf(vals[c] - mNew);
                ls += p;
                sP[wave][r * AKV + c0 + c] = (half_t)p;
            }
            ls += __shfl_xor(ls, 16, 32);
            if (hi == 0) {
                float alpha = __expf(mOld - mNew);
                sM[wave][r]  = mNew;
                sL[wave][r]  = sL[wave][r] * alpha + ls;
                sAl[wave][r] = alpha;
            }
        }
        __syncthreads();

        { // O = O*alpha + P V
            float al[8];
            #pragma unroll
            for (int v = 0; v < 8; ++v) al[v] = sAl[wave][v + hi * 8];
            #pragma unroll
            for (int j = 0; j < 4; ++j)
                #pragma unroll
                for (int v = 0; v < 8; ++v) o[j][v] *= al[v];
            #pragma unroll
            for (int st = 0; st < 2; ++st) {
                union { v16h v; v8h h[2]; } u;
                const half_t* pa = sP[wave] + r * AKV + st * 32 + hi * 16;
                u.h[0] = *(const v8h*)pa;
                u.h[1] = *(const v8h*)(pa + 8);
                v16h pf = u.v;
                #pragma unroll
                for (int j = 0; j < 4; ++j) {
                    v16h vf = *(const v16h*)(sVt + (j * 16 + r) * AKV + st * 32 + hi * 16);
                    o[j] = wmma_f16(pf, vf, o[j]);
                }
            }
        }
        __syncthreads();
    }

    // finalize: O /= l, write [B*S][C] f16, channel = h*64 + d
    float li[8];
    #pragma unroll
    for (int v = 0; v < 8; ++v) li[v] = 1.f / sL[wave][v + hi * 8];
    int b = bh / HH, h = bh % HH;
    #pragma unroll
    for (int j = 0; j < 4; ++j) {
        int d = j * 16 + r;
        #pragma unroll
        for (int v = 0; v < 8; ++v) {
            int q = q0 + v + hi * 8;
            oh[((size_t)(b * SS + q) << 9) + h * DD + d] = (half_t)(o[j][v] * li[v]);
        }
    }
}

// ---------------------------------------------------------------- proj GEMM + bias + residual
__global__ __launch_bounds__(256)
void proj_gemm_kernel(const half_t* __restrict__ A,   // [M][512] attn out
                      const half_t* __restrict__ Bt,  // [512][512] w_proj^T
                      const float* __restrict__ bias,
                      const float* __restrict__ xres,
                      float* __restrict__ out) {
    __shared__ half_t sA[TM * TK];
    __shared__ half_t sB[TN * TK];
    const int tid = threadIdx.x, wave = tid >> 5, lane = tid & 31;
    const int wm = wave & 3, wn = wave >> 2;
    const int m0 = blockIdx.x * TM, n0 = blockIdx.y * TN;
    const int lr = tid >> 1, lk = (tid & 1) * 16;
    const int r = lane & 15, hi = lane >> 4;

    const v8f vz = {0.f,0.f,0.f,0.f,0.f,0.f,0.f,0.f};
    v8f acc[2][4];
    #pragma unroll
    for (int i = 0; i < 2; ++i)
        #pragma unroll
        for (int j = 0; j < 4; ++j) acc[i][j] = vz;

    for (int k0 = 0; k0 < CC; k0 += TK) {
        const half_t* ga = A  + (size_t)(m0 + lr) * CC + k0 + lk;
        const half_t* gb = Bt + (size_t)(n0 + lr) * CC + k0 + lk;
        __builtin_prefetch(ga + TK, 0, 0);
        cp_glds16(sA + lr * TK + lk,     ga);
        cp_glds16(sA + lr * TK + lk + 8, ga + 8);
        cp_glds16(sB + lr * TK + lk,     gb);
        cp_glds16(sB + lr * TK + lk + 8, gb + 8);
        cp_glds_wait();
        __syncthreads();

        v16h af[2], bf[4];
        #pragma unroll
        for (int i = 0; i < 2; ++i) {
            const half_t* p = sA + (wm * 32 + i * 16 + r) * TK + hi * 16;
            union { v16h v; v8h h[2]; } u;
            u.h[0] = *(const v8h*)p;
            u.h[1] = *(const v8h*)(p + 8);
            af[i] = u.v;
        }
        #pragma unroll
        for (int j = 0; j < 4; ++j)
            bf[j] = *(const v16h*)(sB + (wn * 64 + j * 16 + r) * TK + hi * 16);
        #pragma unroll
        for (int i = 0; i < 2; ++i)
            #pragma unroll
            for (int j = 0; j < 4; ++j)
                acc[i][j] = wmma_f16(af[i], bf[j], acc[i][j]);
        __syncthreads();
    }

    #pragma unroll
    for (int i = 0; i < 2; ++i) {
        #pragma unroll
        for (int j = 0; j < 4; ++j) {
            int n = n0 + wn * 64 + j * 16 + r;
            float bi = bias[n];
            #pragma unroll
            for (int v = 0; v < 8; ++v) {
                int m = m0 + wm * 32 + i * 16 + v + hi * 8;
                size_t idx = ((size_t)m << 9) + n;
                out[idx] = acc[i][j][v] + bi + xres[idx];
            }
        }
    }
}

// ---------------------------------------------------------------- launcher
extern "C" void kernel_launch(void* const* d_in, const int* in_sizes, int n_in,
                              void* d_out, int out_size, void* d_ws, size_t ws_size,
                              hipStream_t stream) {
    const float* x      = (const float*)d_in[0];
    const float* gamma  = (const float*)d_in[1];
    const float* beta   = (const float*)d_in[2];
    const float* w_qkv  = (const float*)d_in[3];
    const float* w_proj = (const float*)d_in[4];
    const float* b_proj = (const float*)d_in[5];
    float* out = (float*)d_out;

    char* ws = (char*)d_ws;
    size_t off = 0;
    auto wsalloc = [&](size_t bytes) -> void* {
        void* p = ws + off;
        off = (off + bytes + 255) & ~(size_t)255;
        return p;
    };
    float*  stats   = (float*)wsalloc((size_t)BB * GG * 2 * sizeof(float));
    half_t* xn      = (half_t*)wsalloc((size_t)MT * CC * sizeof(half_t));
    half_t* wqkvT   = (half_t*)wsalloc((size_t)3 * CC * CC * sizeof(half_t));
    half_t* wprojT  = (half_t*)wsalloc((size_t)CC * CC * sizeof(half_t));
    half_t* qh      = (half_t*)wsalloc((size_t)BB * HH * SS * DD * sizeof(half_t));
    half_t* kh      = (half_t*)wsalloc((size_t)BB * HH * SS * DD * sizeof(half_t));
    half_t* vh      = (half_t*)wsalloc((size_t)BB * HH * SS * DD * sizeof(half_t));
    half_t* ao      = (half_t*)wsalloc((size_t)MT * CC * sizeof(half_t));

    gn_stats_kernel<<<BB * GG, 256, 0, stream>>>(x, stats);
    gn_apply_kernel<<<(MT * CC / 4) / 256, 256, 0, stream>>>(x, gamma, beta, stats, xn);
    wt_convert_kernel<<<(CC * 3 * CC) / 256, 256, 0, stream>>>(w_qkv, wqkvT, CC, 3 * CC);
    wt_convert_kernel<<<(CC * CC) / 256, 256, 0, stream>>>(w_proj, wprojT, CC, CC);

    qkv_gemm_kernel<<<dim3(MT / TM, 3 * CC / TN), 256, 0, stream>>>(xn, wqkvT, qh, kh, vh);
    attn_kernel<<<dim3(SS / 128, BB * HH), 256, 0, stream>>>(qh, kh, vh, ao);
    proj_gemm_kernel<<<dim3(MT / TM, CC / TN), 256, 0, stream>>>(ao, wprojT, b_proj, x, out);
}
